// VoteQuery_84756884619595
// MI455X (gfx1250) — matrix operand order repaired and checked by reading, not compiled, not run on hardware
//
#include <hip/hip_runtime.h>
#include <cstdint>

typedef _Float16 half_t;
typedef __attribute__((ext_vector_type(16))) _Float16 v16h;
typedef __attribute__((ext_vector_type(8)))  _Float16 v8h;
typedef __attribute__((ext_vector_type(8)))  float    v8f;

#define B_    16
#define N_    4096
#define C_    256
#define NQ_   256
#define NS_   16
#define T_    (B_*NQ_*NS_)      /* 65536 grouped tokens */
#define KP3   288               /* K=259 padded to 9 K-chunks of 32 */
#define M3P   288               /* M=259 padded to 9 M-blocks of 32 */
#define RAD_  0.3f
#define EPSBN 1e-5f

// ---------------------------------------------------------------------------
// CDNA5 async global->LDS copy (ASYNCcnt tracked), per CDNA5 ISA
// ---------------------------------------------------------------------------
__device__ __forceinline__ void async_b128(uint32_t lds_off, const void* gptr) {
    asm volatile("global_load_async_to_lds_b128 %0, %1, off"
                 :: "v"(lds_off), "v"((uint64_t)(uintptr_t)gptr) : "memory");
}
__device__ __forceinline__ void wait_async0() {
    asm volatile("s_wait_asynccnt 0x0" ::: "memory");
}

#define SHUF16(lo, hi) __builtin_shufflevector(lo, hi, 0,1,2,3,4,5,6,7,8,9,10,11,12,13,14,15)

// ---------------------------------------------------------------------------
// WMMA GEMM:  Y(b, M, n) = A(M,Kpad) * X(b, n, Kpad)^T + bias(M)
//  A: f16 row-major (lda = Kpad), zero-padded rows/cols.
//  X: f16 TOKEN-major per batch (token stride = Kpad), zero-padded K.
//  Block = 256 thr (8 waves) -> 32(M) x 128(N) tile; wave = two 16x16 tiles.
//  Per K-chunk: async-stage A(32x32) + X(128x32) into LDS, fragments are
//  pure ds_load_b128 (token-major makes B contiguous; A runs are contiguous
//  per the ISA 16-bit A layout: K = 8h..8h+7 and 16+8h..23+8h).
// ---------------------------------------------------------------------------
__global__ __launch_bounds__(256)
void gemm_wmma(const half_t* __restrict__ A, const half_t* __restrict__ X,
               const float* __restrict__ bias, float* __restrict__ Y,
               int Mreal, int Kpad, long Ncols, long bsX, long bsY)
{
    __shared__ half_t lA[32 * 32];     // 2 KB : 32 M-rows x 32 K
    __shared__ half_t lX[128 * 32];    // 8 KB : 128 tokens x 32 K
    const int tid  = threadIdx.x;
    const int lane = tid & 31;
    const int w    = tid >> 5;         // wave id 0..7
    const int h    = lane >> 4;        // half-wave
    const int l16  = lane & 15;
    const long n0  = (long)blockIdx.x * 128;
    const int  mo  = blockIdx.y;       // 32-row M block
    const long b   = blockIdx.z;
    const half_t* Xb = X + b * bsX;

    // staging assignments (base pointers; advance by k0 halves each chunk)
    const int arow = tid >> 2, aseg = tid & 3;                    // tid < 128
    const half_t* gA = A + (long)(mo * 32 + arow) * Kpad + aseg * 8;
    const uint32_t lAo = (uint32_t)(uintptr_t)&lA[arow * 32 + aseg * 8];
    const int li1 = tid + 256;
    const half_t* gX0 = Xb + (long)(n0 + (tid >> 2)) * Kpad + (tid & 3) * 8;
    const half_t* gX1 = Xb + (long)(n0 + (li1 >> 2)) * Kpad + (li1 & 3) * 8;
    const uint32_t lX0 = (uint32_t)(uintptr_t)&lX[(tid >> 2) * 32 + (tid & 3) * 8];
    const uint32_t lX1 = (uint32_t)(uintptr_t)&lX[(li1 >> 2) * 32 + (li1 & 3) * 8];

    v8f c0 = {}, c1 = {};
    for (int k0 = 0; k0 < Kpad; k0 += 32) {
        if (tid < 128) async_b128(lAo, gA + k0);
        async_b128(lX0, gX0 + k0);
        async_b128(lX1, gX1 + k0);
        if (k0 + 32 < Kpad) __builtin_prefetch(gX0 + k0 + 32, 0, 1);
        wait_async0();
        __syncthreads();

        const v8h a0lo = *(const v8h*)&lA[l16 * 32 + 8 * h];
        const v8h a0hi = *(const v8h*)&lA[l16 * 32 + 16 + 8 * h];
        const v8h a1lo = *(const v8h*)&lA[(16 + l16) * 32 + 8 * h];
        const v8h a1hi = *(const v8h*)&lA[(16 + l16) * 32 + 16 + 8 * h];
        const v8h blo  = *(const v8h*)&lX[(w * 16 + l16) * 32 + 8 * h];
        const v8h bhi  = *(const v8h*)&lX[(w * 16 + l16) * 32 + 16 + 8 * h];
        v16h a0 = SHUF16(a0lo, a0hi);
        v16h a1 = SHUF16(a1lo, a1hi);
        v16h bm = SHUF16(blo,  bhi);
        c0 = __builtin_amdgcn_wmma_f32_16x16x32_f16(false, a0, false, bm, (short)0, c0, false, false);
        c1 = __builtin_amdgcn_wmma_f32_16x16x32_f16(false, a1, false, bm, (short)0, c1, false, false);
        __syncthreads();
    }
    // C/D layout: VGPR r, lane l -> M = r + 8*(l/16), N = l%16
    #pragma unroll
    for (int r = 0; r < 8; ++r) {
        long ncol = n0 + w * 16 + l16;
        int M0 = mo * 32 + r + 8 * h;
        int M1 = M0 + 16;
        if (M0 < Mreal) Y[b * bsY + (long)M0 * Ncols + ncol] = c0[r] + bias[M0];
        if (M1 < Mreal) Y[b * bsY + (long)M1 * Ncols + ncol] = c1[r] + bias[M1];
    }
}

// ---------------------------------------------------------------------------
// elementwise helpers
// ---------------------------------------------------------------------------
// (B,C,N) f32 -> token-major (B,N,C) f16
__global__ void cvt_act_tr(const float* __restrict__ x, half_t* __restrict__ xh, long n) {
    long i = (long)blockIdx.x * 256 + threadIdx.x;
    if (i >= n) return;
    long b = i / ((long)N_ * C_);
    long r = i % ((long)N_ * C_);
    long nn = r / C_;
    int  c  = (int)(r % C_);
    xh[i] = (half_t)x[b * (long)C_ * N_ + (long)c * N_ + nn];
}

__global__ void cvt_weight(const float* __restrict__ W, half_t* __restrict__ Wh,
                           int O, int K, int Opad, int Kpad) {
    long i = (long)blockIdx.x * 256 + threadIdx.x;
    if (i >= (long)Opad * Kpad) return;
    int o = (int)(i / Kpad), k = (int)(i % Kpad);
    Wh[i] = (o < O && k < K) ? (half_t)W[(long)o * K + k] : (half_t)0.f;
}

__global__ __launch_bounds__(256)
void bn_stats(const float* __restrict__ Y, float* __restrict__ mean,
              float* __restrict__ var, long Ncols, int Bn, long bsY) {
    int o = blockIdx.x;
    __shared__ float ss[256], ss2[256];
    float s = 0.f, s2 = 0.f;
    long total = (long)Bn * Ncols;
    for (long i = threadIdx.x; i < total; i += 256) {
        long bb = i / Ncols, n = i % Ncols;
        float v = Y[bb * bsY + (long)o * Ncols + n];
        s += v; s2 += v * v;
    }
    ss[threadIdx.x] = s; ss2[threadIdx.x] = s2;
    __syncthreads();
    for (int st = 128; st > 0; st >>= 1) {
        if (threadIdx.x < st) {
            ss[threadIdx.x]  += ss[threadIdx.x + st];
            ss2[threadIdx.x] += ss2[threadIdx.x + st];
        }
        __syncthreads();
    }
    if (threadIdx.x == 0) {
        float m = ss[0] / (float)total;
        mean[o] = m;
        var[o]  = ss2[0] / (float)total - m * m;
    }
}

// Y (b, O, n) f32 -> BN+ReLU -> token-major f16 out[bb*bsOut + n*O + o]
__global__ void bn_relu_cvt(const float* __restrict__ Y, const float* __restrict__ mean,
                            const float* __restrict__ var, const float* __restrict__ g,
                            const float* __restrict__ be, half_t* __restrict__ out,
                            int O, long Ncols, long bsY, long bsOut, long total) {
    long i = (long)blockIdx.x * 256 + threadIdx.x;
    if (i >= total) return;
    long per = (long)O * Ncols;
    long bb = i / per, r = i % per;
    int  o  = (int)(r / Ncols);
    long n  = r % Ncols;
    float v = Y[bb * bsY + (long)o * Ncols + n];
    v = g[o] * (v - mean[o]) * rsqrtf(var[o] + EPSBN) + be[o];
    v = v > 0.f ? v : 0.f;
    out[bb * bsOut + n * (long)O + o] = (half_t)v;
}

// vote_xyz = encode_xyz + clip(sigmoid(Y3[0:3]),0.1,0.9) - 0.5 ; also copy xyz
__global__ void vote_kernel(const float* __restrict__ Y3, const float* __restrict__ exyz,
                            float* __restrict__ vout, float* __restrict__ excopy, long bsY) {
    long i = (long)blockIdx.x * 256 + threadIdx.x;
    if (i >= (long)B_ * N_) return;
    long b = i / N_, n = i % N_;
    #pragma unroll
    for (int d = 0; d < 3; ++d) {
        float y = Y3[b * bsY + (long)d * N_ + n];
        float s = 1.f / (1.f + expf(-y));
        s = fminf(fmaxf(s, 0.1f), 0.9f);
        float e = exyz[(b * N_ + n) * 3 + d];
        vout[(b * N_ + n) * 3 + d]   = e + s - 0.5f;
        excopy[(b * N_ + n) * 3 + d] = e;
    }
}

// feat = L2-normalize_C(encode_features + Y3[3:259]) -> token-major f16 (B,N,C)
__global__ void feat_kernel(const float* __restrict__ ef, const float* __restrict__ Y3,
                            half_t* __restrict__ feath, long bsY) {
    long i = (long)blockIdx.x * 256 + threadIdx.x;
    if (i >= (long)B_ * N_) return;
    long b = i / N_, n = i % N_;
    float ss = 0.f;
    for (int c = 0; c < C_; ++c) {
        float v = ef[b * (long)C_ * N_ + (long)c * N_ + n] + Y3[b * bsY + (long)(3 + c) * N_ + n];
        ss += v * v;
    }
    float inv = rsqrtf(ss);
    for (int c = 0; c < C_; ++c) {
        float v = ef[b * (long)C_ * N_ + (long)c * N_ + n] + Y3[b * bsY + (long)(3 + c) * N_ + n];
        feath[(b * N_ + n) * C_ + c] = (half_t)(v * inv);
    }
}

// sequential FPS: one block per batch, 256 threads x 16 points each
__global__ __launch_bounds__(256)
void fps_kernel(const float* __restrict__ xyz, int* __restrict__ sinds,
                float* __restrict__ sindf) {
    int b = blockIdx.x;
    const float* P = xyz + (long)b * N_ * 3;
    float dist[16];
    #pragma unroll
    for (int j = 0; j < 16; ++j) dist[j] = 1e10f;
    __shared__ float smax[256];
    __shared__ int   sarg[256];
    __shared__ int   sfar;
    int far = 0;
    for (int it = 0; it < NQ_; ++it) {
        if (threadIdx.x == 0) { sinds[b * NQ_ + it] = far; sindf[b * NQ_ + it] = (float)far; }
        float cx = P[far * 3], cy = P[far * 3 + 1], cz = P[far * 3 + 2];
        float bm = -1.f; int ba = 0;
        #pragma unroll
        for (int j = 0; j < 16; ++j) {
            int n = threadIdx.x + j * 256;
            float dx = P[n * 3] - cx, dy = P[n * 3 + 1] - cy, dz = P[n * 3 + 2] - cz;
            float d = dx * dx + dy * dy + dz * dz;
            dist[j] = fminf(dist[j], d);
            if (dist[j] > bm) { bm = dist[j]; ba = n; }
        }
        smax[threadIdx.x] = bm; sarg[threadIdx.x] = ba;
        __syncthreads();
        for (int st = 128; st > 0; st >>= 1) {
            if (threadIdx.x < st) {
                if (smax[threadIdx.x + st] > smax[threadIdx.x] ||
                    (smax[threadIdx.x + st] == smax[threadIdx.x] &&
                     sarg[threadIdx.x + st] < sarg[threadIdx.x])) {
                    smax[threadIdx.x] = smax[threadIdx.x + st];
                    sarg[threadIdx.x] = sarg[threadIdx.x + st];
                }
            }
            __syncthreads();
        }
        if (threadIdx.x == 0) sfar = sarg[0];
        __syncthreads();
        far = sfar;
        __syncthreads();
    }
}

__global__ void qxyz_kernel(const float* __restrict__ vote, const int* __restrict__ sinds,
                            float* __restrict__ qxyz) {
    long i = (long)blockIdx.x * 256 + threadIdx.x;
    if (i >= (long)B_ * NQ_) return;
    long b = i / NQ_;
    int  n = sinds[i];
    #pragma unroll
    for (int d = 0; d < 3; ++d)
        qxyz[i * 3 + d] = vote[(b * N_ + n) * 3 + d];
}

// ball query: ascending-index neighbors within radius, fill with first
__global__ void ballq_kernel(const float* __restrict__ vote, const float* __restrict__ qxyz,
                             int* __restrict__ idx) {
    long i = (long)blockIdx.x * 256 + threadIdx.x;
    if (i >= (long)B_ * NQ_) return;
    long b = i / NQ_;
    const float* V = vote + b * (long)N_ * 3;
    float qx = qxyz[i * 3], qy = qxyz[i * 3 + 1], qz = qxyz[i * 3 + 2];
    int cnt = 0, first = 0, out[NS_];
    for (int n = 0; n < N_ && cnt < NS_; ++n) {
        float dx = V[n * 3] - qx, dy = V[n * 3 + 1] - qy, dz = V[n * 3 + 2] - qz;
        if (dx * dx + dy * dy + dz * dz < RAD_ * RAD_) {
            if (cnt == 0) first = n;
            out[cnt++] = n;
        }
    }
    for (int s = 0; s < NS_; ++s) idx[i * NS_ + s] = (s < cnt) ? out[s] : first;
}

// build grouped tensor Gh token-major (T, KP3): [xyz/R | feat | zero-pad]
__global__ void gather_kernel(const float* __restrict__ vote, const float* __restrict__ qxyz,
                              const int* __restrict__ idx, const half_t* __restrict__ feath,
                              half_t* __restrict__ Gh) {
    long t = (long)blockIdx.x * 256 + threadIdx.x;
    if (t >= (long)T_) return;
    long b  = t / (NQ_ * NS_);
    int rem = (int)(t % (NQ_ * NS_));
    int q   = rem / NS_;
    int n   = idx[t];
    long bq = b * NQ_ + q;
    #pragma unroll
    for (int d = 0; d < 3; ++d)
        Gh[t * KP3 + d] =
            (half_t)((vote[(b * N_ + n) * 3 + d] - qxyz[bq * 3 + d]) * (1.f / RAD_));
    for (int c = 0; c < C_; ++c)
        Gh[t * KP3 + 3 + c] = feath[(b * N_ + n) * C_ + c];
    for (int c = C_ + 3; c < KP3; ++c)
        Gh[t * KP3 + c] = (half_t)0.f;
}

// query_features[b,o,q] = max over s ; H token-major (T, C)
__global__ void maxpool_kernel(const half_t* __restrict__ H, float* __restrict__ out) {
    long i = (long)blockIdx.x * 256 + threadIdx.x;
    if (i >= (long)B_ * C_ * NQ_) return;
    long b  = i / (C_ * NQ_);
    int rem = (int)(i % (C_ * NQ_));
    int o = rem / NQ_, q = rem % NQ_;
    long tb = (b * NQ_ + q) * NS_;
    float m = 0.f;   // post-ReLU values are >= 0
    for (int s = 0; s < NS_; ++s) m = fmaxf(m, (float)H[(tb + s) * C_ + o]);
    out[i] = m;
}

// ---------------------------------------------------------------------------
extern "C" void kernel_launch(void* const* d_in, const int* in_sizes, int n_in,
                              void* d_out, int out_size, void* d_ws, size_t ws_size,
                              hipStream_t stream) {
    const float* exyz = (const float*)d_in[0];
    const float* efeat= (const float*)d_in[1];
    const float* W1 = (const float*)d_in[2];  const float* b1 = (const float*)d_in[3];
    const float* g1 = (const float*)d_in[4];  const float* be1= (const float*)d_in[5];
    const float* W2 = (const float*)d_in[6];  const float* b2 = (const float*)d_in[7];
    const float* g2 = (const float*)d_in[8];  const float* be2= (const float*)d_in[9];
    const float* W3 = (const float*)d_in[10]; const float* b3 = (const float*)d_in[11];
    const float* mW1= (const float*)d_in[12]; const float* mb1= (const float*)d_in[13];
    const float* mg1= (const float*)d_in[14]; const float* mbe1=(const float*)d_in[15];
    const float* mW2= (const float*)d_in[16]; const float* mb2= (const float*)d_in[17];
    const float* mg2= (const float*)d_in[18]; const float* mbe2=(const float*)d_in[19];
    const float* mW3= (const float*)d_in[20]; const float* mb3= (const float*)d_in[21];
    const float* mg3= (const float*)d_in[22]; const float* mbe3=(const float*)d_in[23];

    char* ws = (char*)d_ws;
    half_t* Xh   = (half_t*)(ws + 0);                 // 32 MB ping (token-major)
    half_t* Bufh = (half_t*)(ws + 33554432);          // 32 MB pong (token-major)
    float*  Ybuf = (float* )(ws + 67108864);          // 75.5 MB GEMM out (B,288,N)
    half_t* Gh   = (half_t*)(ws + 142606336);         // 36 MB grouped (T,288)
    half_t* W1h  = (half_t*)(ws + 180355072);
    half_t* W2h  = (half_t*)(ws + 180486144);
    half_t* W3h  = (half_t*)(ws + 180617216);         // 288x256
    half_t* mW1h = (half_t*)(ws + 180764672);         // 256x288
    half_t* mW2h = (half_t*)(ws + 180912128);
    half_t* mW3h = (half_t*)(ws + 181043200);
    float*  mean = (float* )(ws + 181174272);
    float*  var  = (float* )(ws + 181175424);
    int*    sind = (int*   )(ws + 181176576);
    int*    bidx = (int*   )(ws + 181192960);

    float* out   = (float*)d_out;
    float* vote  = out;                    // (B,N,3)
    float* exout = out + 196608;           // (B,N,3)
    float* sindf = out + 393216;           // (B,NQ)
    float* qxyz  = out + 397312;           // (B,NQ,3)
    float* qfeat = out + 409600;           // (B,C,NQ)

    const long bsXf = (long)N_ * C_;       // f16 token-major batch stride (FFN)
    const long bsYf = (long)M3P * N_;      // Ybuf batch stride (FFN)
    const long totF = (long)B_ * C_ * N_;  // 16.7M
    const long totS = (long)C_ * T_;       // 16.7M

    // --- f32 -> f16 conversions (once) ------------------------------------
    cvt_act_tr<<<(totF + 255) / 256, 256, 0, stream>>>(efeat, Xh, totF);
    cvt_weight<<<(256 * 256 + 255) / 256, 256, 0, stream>>>(W1, W1h, 256, 256, 256, 256);
    cvt_weight<<<(256 * 256 + 255) / 256, 256, 0, stream>>>(W2, W2h, 256, 256, 256, 256);
    cvt_weight<<<(M3P * 256 + 255) / 256, 256, 0, stream>>>(W3, W3h, 259, 256, M3P, 256);
    cvt_weight<<<(256 * KP3 + 255) / 256, 256, 0, stream>>>(mW1, mW1h, 256, 259, 256, KP3);
    cvt_weight<<<(256 * 256 + 255) / 256, 256, 0, stream>>>(mW2, mW2h, 256, 256, 256, 256);
    cvt_weight<<<(256 * 256 + 255) / 256, 256, 0, stream>>>(mW3, mW3h, 256, 256, 256, 256);

    // --- FFN_vote ----------------------------------------------------------
    dim3 gF(32, 8, B_);
    gemm_wmma<<<gF, 256, 0, stream>>>(W1h, Xh, b1, Ybuf, 256, 256, N_, bsXf, bsYf);
    bn_stats<<<256, 256, 0, stream>>>(Ybuf, mean, var, N_, B_, bsYf);
    bn_relu_cvt<<<(totF + 255) / 256, 256, 0, stream>>>(Ybuf, mean, var, g1, be1, Bufh,
                                                        256, N_, bsYf, bsXf, totF);
    gemm_wmma<<<gF, 256, 0, stream>>>(W2h, Bufh, b2, Ybuf, 256, 256, N_, bsXf, bsYf);
    bn_stats<<<256, 256, 0, stream>>>(Ybuf, mean, var, N_, B_, bsYf);
    bn_relu_cvt<<<(totF + 255) / 256, 256, 0, stream>>>(Ybuf, mean, var, g2, be2, Xh,
                                                        256, N_, bsYf, bsXf, totF);
    dim3 g3(32, 9, B_);
    gemm_wmma<<<g3, 256, 0, stream>>>(W3h, Xh, b3, Ybuf, 259, 256, N_, bsXf, bsYf);

    // --- vote / normalized features ---------------------------------------
    long bn = (long)B_ * N_;
    vote_kernel<<<(bn + 255) / 256, 256, 0, stream>>>(Ybuf, exyz, vote, exout, bsYf);
    feat_kernel<<<(bn + 255) / 256, 256, 0, stream>>>(efeat, Ybuf, Bufh, bsYf);

    // --- FPS / query / ball query / gather --------------------------------
    fps_kernel<<<B_, 256, 0, stream>>>(exyz, sind, sindf);
    qxyz_kernel<<<(B_ * NQ_ + 255) / 256, 256, 0, stream>>>(vote, sind, qxyz);
    ballq_kernel<<<(B_ * NQ_ + 255) / 256, 256, 0, stream>>>(vote, qxyz, bidx);
    gather_kernel<<<(T_ + 255) / 256, 256, 0, stream>>>(vote, qxyz, bidx, Bufh, Gh);

    // --- SA shared MLP ------------------------------------------------------
    dim3 gS(512, 8, 1);
    gemm_wmma<<<gS, 256, 0, stream>>>(mW1h, Gh, mb1, Ybuf, 256, KP3, T_, 0, 0);
    bn_stats<<<256, 256, 0, stream>>>(Ybuf, mean, var, T_, 1, 0);
    bn_relu_cvt<<<(totS + 255) / 256, 256, 0, stream>>>(Ybuf, mean, var, mg1, mbe1, Xh,
                                                        256, T_, 0, 0, totS);
    gemm_wmma<<<gS, 256, 0, stream>>>(mW2h, Xh, mb2, Ybuf, 256, 256, T_, 0, 0);
    bn_stats<<<256, 256, 0, stream>>>(Ybuf, mean, var, T_, 1, 0);
    bn_relu_cvt<<<(totS + 255) / 256, 256, 0, stream>>>(Ybuf, mean, var, mg2, mbe2, Gh,
                                                        256, T_, 0, 0, totS);
    gemm_wmma<<<gS, 256, 0, stream>>>(mW3h, Gh, mb3, Ybuf, 256, 256, T_, 0, 0);
    bn_stats<<<256, 256, 0, stream>>>(Ybuf, mean, var, T_, 1, 0);
    bn_relu_cvt<<<(totS + 255) / 256, 256, 0, stream>>>(Ybuf, mean, var, mg3, mbe3, Xh,
                                                        256, T_, 0, 0, totS);

    // --- max over samples ---------------------------------------------------
    long bq = (long)B_ * C_ * NQ_;
    maxpool_kernel<<<(bq + 255) / 256, 256, 0, stream>>>(Xh, qfeat);
}
